// AF_CILN_15951508538089
// MI455X (gfx1250) — matrix operand order repaired and checked by reading, not compiled, not run on hardware
//
#include <hip/hip_runtime.h>
#include <hip/hip_bf16.h>

// ---------------------------------------------------------------------------
// Problem constants (from reference)
// ---------------------------------------------------------------------------
constexpr int nB = 32, nL = 1024, nC = 128, nP = 96, nK = 5, NNODES = 15;

typedef __attribute__((ext_vector_type(16))) _Float16 v16h;
typedef __attribute__((ext_vector_type(8)))  float    v8f;
typedef __attribute__((ext_vector_type(2)))  _Float16 h2;

// ---------------------------------------------------------------------------
// CDNA5 async global->LDS copy (ASYNCcnt-tracked, VGPR-bypassing)
// ---------------------------------------------------------------------------
__device__ __forceinline__ void async_copy_b128(void* lds_dst, const void* gsrc) {
  unsigned lds = (unsigned)(unsigned long long)lds_dst;   // low 32 bits = LDS offset
  asm volatile("global_load_async_to_lds_b128 %0, %1, off"
               :: "v"(lds), "v"(gsrc) : "memory");
}
__device__ __forceinline__ void wait_async() {
  asm volatile("s_wait_asynccnt 0x0" ::: "memory");
}

// ---------------------------------------------------------------------------
// RevIN statistics: mean / std per (b, c)  (mask-aware, wave32 shuffle reduce)
// ---------------------------------------------------------------------------
__global__ void __launch_bounds__(256) k_norm(const float* __restrict__ x,
                                              const float* __restrict__ mask,
                                              float* __restrict__ means,
                                              float* __restrict__ stdev) {
  int row  = blockIdx.x * 8 + (threadIdx.x >> 5);   // b*C + c
  int lane = threadIdx.x & 31;
  int b = row >> 7, c = row & 127;
  const float* xp = x    + (size_t)b * nL * nC + c;
  const float* mp = mask + (size_t)b * nL * nC + c;
  float s = 0.f, cnt = 0.f;
  for (int l = lane; l < nL; l += 32) {
    s   += xp[(size_t)l * nC];
    cnt += (mp[(size_t)l * nC] == 1.0f) ? 1.0f : 0.0f;
  }
  for (int off = 16; off; off >>= 1) { s += __shfl_xor(s, off, 32); cnt += __shfl_xor(cnt, off, 32); }
  float mean = s / cnt;
  float ss = 0.f;
  for (int l = lane; l < nL; l += 32) {
    float m = mp[(size_t)l * nC];
    float v = (m == 0.0f) ? 0.0f : (xp[(size_t)l * nC] - mean);
    ss += v * v;
  }
  for (int off = 16; off; off >>= 1) ss += __shfl_xor(ss, off, 32);
  if (lane == 0) { means[row] = mean; stdev[row] = sqrtf(ss / cnt + 1e-5f); }
}

__global__ void __launch_bounds__(256) k_prep(const float* __restrict__ x,
                                              const float* __restrict__ mask,
                                              const float* __restrict__ means,
                                              const float* __restrict__ stdev,
                                              float* __restrict__ xn) {
  size_t i = (size_t)blockIdx.x * 256 + threadIdx.x;   // over B*L*C
  int c = (int)(i & 127);
  size_t bl = i >> 7;
  int b = (int)(bl >> 10);
  float m = mask[i];
  float v = (m == 0.f) ? 0.f : (x[i] - means[b * nC + c]);
  xn[i] = v / stdev[b * nC + c];
}

// ---------------------------------------------------------------------------
// Haar wavelet over L=1024 per (b, c) row; outputs f16 in reference ordering
// ---------------------------------------------------------------------------
__global__ void __launch_bounds__(256) k_haar(const float* __restrict__ xn,
                                              _Float16* __restrict__ freq) {
  __shared__ float buf[nL];
  int row = blockIdx.x;                 // b*C + c
  int b = row >> 7, c = row & 127;
  int tid = threadIdx.x;
  for (int l = tid; l < nL; l += 256)
    buf[l] = xn[((size_t)b * nL + l) * nC + c];
  __syncthreads();
  const float is2 = 0.70710678118654752440f;
  for (int nh = 512; nh >= 1; nh >>= 1) {
    float e0 = 0.f, o0 = 0.f, e1 = 0.f, o1 = 0.f;
    int i0 = tid, i1 = tid + 256;
    if (i0 < nh) { e0 = buf[2 * i0]; o0 = buf[2 * i0 + 1]; }
    if (i1 < nh) { e1 = buf[2 * i1]; o1 = buf[2 * i1 + 1]; }
    __syncthreads();
    if (i0 < nh) {
      buf[i0] = (e0 + o0) * is2;
      freq[(size_t)row * nL + nh + i0] = (_Float16)((e0 - o0) * is2);
    }
    if (i1 < nh) {
      buf[i1] = (e1 + o1) * is2;
      freq[(size_t)row * nL + nh + i1] = (_Float16)((e1 - o1) * is2);
    }
    __syncthreads();
  }
  if (tid == 0) freq[(size_t)row * nL] = (_Float16)buf[0];
}

// ---------------------------------------------------------------------------
// f32 -> f16 converters (weights)
// ---------------------------------------------------------------------------
__global__ void __launch_bounds__(256) k_cvt(const float* __restrict__ s,
                                             _Float16* __restrict__ d, int n) {
  int i = blockIdx.x * 256 + threadIdx.x;
  if (i < n) d[i] = (_Float16)s[i];
}

// conv weights: src (NN,4,Cout,Cin,K) -> dst (NN,4,Cout,K,Cin)
__global__ void __launch_bounds__(256) k_cvtw(const float* __restrict__ s,
                                              _Float16* __restrict__ d, int n) {
  int i = blockIdx.x * 256 + threadIdx.x;
  if (i >= n) return;
  int k = i % nK; int t = i / nK;
  int cin = t % nC; t /= nC;
  int cout = t % nC; int nf = t / nC;
  d[(((size_t)nf * nC + cout) * nK + k) * nC + cin] = (_Float16)s[i];
}

// ---------------------------------------------------------------------------
// f16 WMMA GEMM:  out[m][n] = act( sum_k A[m][k] * W[n][k] )
// Block = 8 waves -> 128x128 tile. A/B k-slices staged in LDS via async copy.
// ACT 0: ReLU -> f16 out; ACT 1: sigmoid -> f32 out
// ---------------------------------------------------------------------------
template <int ACT>
__global__ void __launch_bounds__(256) k_gemm(const _Float16* __restrict__ A,
                                              const _Float16* __restrict__ W,
                                              void* __restrict__ out,
                                              int M, int N, int Kd) {
  __shared__ _Float16 As[128 * 32];
  __shared__ _Float16 Bs[128 * 32];
  int mb = blockIdx.x * 128, nb = blockIdx.y * 128;
  int tid = threadIdx.x;
  int wave = tid >> 5, lane = tid & 31;
  int n = lane & 15, h = lane >> 4;
  v8f acc[8];
  #pragma unroll
  for (int j = 0; j < 8; ++j) acc[j] = (v8f){};

  for (int k0 = 0; k0 < Kd; k0 += 32) {
    // Stage 128x32 A and B slices: 512 x 16B chunks each, 2 per thread each.
    #pragma unroll
    for (int j = 0; j < 2; ++j) {
      int idx = tid * 2 + j;                 // 0..511
      int row = idx >> 2, q = idx & 3;       // q: which 16B of the 64B row
      async_copy_b128(&As[row * 32 + q * 8], A + (size_t)(mb + row) * Kd + k0 + q * 8);
      async_copy_b128(&Bs[row * 32 + q * 8], W + (size_t)(nb + row) * Kd + k0 + q * 8);
    }
    wait_async();
    __syncthreads();

    v16h a;
    const _Float16* ar = &As[(wave * 16 + (lane & 15)) * 32];
    #pragma unroll
    for (int v = 0; v < 8; ++v) {
      int Ko = (v < 4) ? (h * 8 + 2 * v) : (16 + h * 8 + 2 * (v - 4));
      a[2 * v] = ar[Ko]; a[2 * v + 1] = ar[Ko + 1];
    }
    #pragma unroll
    for (int j = 0; j < 8; ++j) {
      const _Float16* br = &Bs[(j * 16 + n) * 32];
      v16h bf;
      #pragma unroll
      for (int v = 0; v < 8; ++v) {
        int Ko = (v < 4) ? (h * 8 + 2 * v) : (16 + h * 8 + 2 * (v - 4));
        bf[2 * v] = br[Ko]; bf[2 * v + 1] = br[Ko + 1];
      }
      acc[j] = __builtin_amdgcn_wmma_f32_16x16x32_f16(false, a, false, bf,
                                                      (short)0, acc[j], false, false);
    }
    __syncthreads();   // protect LDS before next k-step overwrites
  }

  #pragma unroll
  for (int j = 0; j < 8; ++j) {
    #pragma unroll
    for (int i = 0; i < 8; ++i) {
      int r = mb + wave * 16 + h * 8 + i;
      int c = nb + j * 16 + n;
      float v = acc[j][i];
      if (ACT == 0) {
        v = v > 0.f ? v : 0.f;
        ((_Float16*)out)[(size_t)r * N + c] = (_Float16)v;
      } else {
        v = 1.0f / (1.0f + __expf(-v));
        ((float*)out)[(size_t)r * N + c] = v;
      }
    }
  }
}

// ---------------------------------------------------------------------------
// LayerNorm over last dim (1024) of lr, in place
// ---------------------------------------------------------------------------
__global__ void __launch_bounds__(256) k_ln(float* __restrict__ lr,
                                            const float* __restrict__ g,
                                            const float* __restrict__ bb) {
  __shared__ float red[8];
  int row = blockIdx.x;
  float* p = lr + (size_t)row * nL;
  int tid = threadIdx.x;
  float v[4]; float s = 0.f;
  #pragma unroll
  for (int j = 0; j < 4; ++j) { v[j] = p[tid + 256 * j]; s += v[j]; }
  for (int off = 16; off; off >>= 1) s += __shfl_xor(s, off, 32);
  if ((tid & 31) == 0) red[tid >> 5] = s;
  __syncthreads();
  s = 0.f;
  #pragma unroll
  for (int j = 0; j < 8; ++j) s += red[j];
  float mu = s * (1.0f / 1024.0f);
  __syncthreads();
  float q = 0.f;
  #pragma unroll
  for (int j = 0; j < 4; ++j) { float d = v[j] - mu; q += d * d; }
  for (int off = 16; off; off >>= 1) q += __shfl_xor(q, off, 32);
  if ((tid & 31) == 0) red[tid >> 5] = q;
  __syncthreads();
  q = 0.f;
  #pragma unroll
  for (int j = 0; j < 8; ++j) q += red[j];
  float inv = rsqrtf(q * (1.0f / 1024.0f) + 1e-6f);
  #pragma unroll
  for (int j = 0; j < 4; ++j) {
    int l = tid + 256 * j;
    p[l] = (v[j] - mu) * inv * g[l] + bb[l];
  }
}

// x0[b][l][c] = xn * lr[b][c][l] + pe[l][c]
__global__ void __launch_bounds__(256) k_apply(const float* __restrict__ xn,
                                               const float* __restrict__ lr,
                                               float* __restrict__ x0) {
  size_t i = (size_t)blockIdx.x * 256 + threadIdx.x;
  int c = (int)(i & 127);
  size_t bl = i >> 7;
  int l = (int)(bl & 1023);
  int b = (int)(bl >> 10);
  float scale = lr[((size_t)b * nC + c) * nL + l];
  const float inc = 0.146195879804146f;   // log(10000)/63
  int ti = c & 63;
  float st = (float)l * __expf(-inc * (float)ti);
  float pe = (c < 64) ? __sinf(st) : __cosf(st);
  x0[i] = xn[i] * scale + pe;
}

// ---------------------------------------------------------------------------
// Tree: split even/odd rows, transpose to (B,C,T), edge-pad K-1=4, f16 copy
// ---------------------------------------------------------------------------
__global__ void __launch_bounds__(256) k_split(const float* __restrict__ zin, int Lz,
                                               float* __restrict__ xe, float* __restrict__ xo,
                                               _Float16* __restrict__ xep,
                                               _Float16* __restrict__ xop, int T) {
  int Tp = T + 8;
  size_t i = (size_t)blockIdx.x * 256 + threadIdx.x;
  if (i >= (size_t)nB * nC * Tp) return;
  int tp = (int)(i % Tp);
  size_t t2 = i / Tp;
  int ci = (int)(t2 % nC);
  int b  = (int)(t2 / nC);
  int t = tp - 4;
  t = t < 0 ? 0 : (t >= T ? T - 1 : t);
  float e = zin[((size_t)b * Lz + 2 * t)     * nC + ci];
  float o = zin[((size_t)b * Lz + 2 * t + 1) * nC + ci];
  xep[i] = (_Float16)e;
  xop[i] = (_Float16)o;
  if (tp >= 4 && tp < T + 4) {
    size_t j = ((size_t)b * nC + ci) * T + (tp - 4);
    xe[j] = e; xo[j] = o;
  }
}

// ---------------------------------------------------------------------------
// Causal conv block:  y = tanh( conv2( lrelu( conv1(xpad)+b1 ) ) + b2 )
// Implicit-GEMM via WMMA 16x16x32 f16. Block = 8 waves, 64 output cols.
// Input tile async-copied to LDS (contiguous b128), then transposed in LDS.
// Weights laid out [cout][k][cin] (K-dim contiguous), L2-resident.
// ---------------------------------------------------------------------------
__global__ void __launch_bounds__(256) k_ccb(const _Float16* __restrict__ xpad,
                                             const _Float16* __restrict__ wA,
                                             const float* __restrict__ b1,
                                             const _Float16* __restrict__ wB,
                                             const float* __restrict__ b2,
                                             float* __restrict__ y, int T, int Tp) {
  __shared__ _Float16 Xraw[128 * 80];  // [ci][t 0..71], row stride 80
  __shared__ _Float16 Xs[72 * nC];     // [t][ci]
  __shared__ _Float16 Ms[80 * nC];     // [m][cout]
  int b    = blockIdx.x;
  int t0   = blockIdx.y * 64;
  int tid  = threadIdx.x;

  // warm L2/L0 for the weight streams (shared by all blocks)
  __builtin_prefetch(wA + (size_t)tid * 256, 0, 3);
  __builtin_prefetch(wB + (size_t)tid * 256, 0, 3);

  // async-stage the input tile: 128 rows x 72 f16 = 9 x 16B chunks per row
  for (int idx = tid; idx < 128 * 9; idx += 256) {
    int row = idx / 9, q = idx % 9;
    async_copy_b128(&Xraw[row * 80 + q * 8],
                    xpad + ((size_t)b * nC + row) * Tp + t0 + q * 8);
  }
  wait_async();
  __syncthreads();
  // transpose [ci][t] -> [t][ci] inside LDS
  for (int idx = tid; idx < 72 * nC; idx += 256) {
    int t = idx >> 7, ci = idx & 127;
    Xs[idx] = Xraw[ci * 80 + t];
  }
  __syncthreads();

  int wave = tid >> 5, lane = tid & 31;
  int n = lane & 15, h = lane >> 4;

  // conv1 -> Mid (cols 0..79; only 0..67 meaningful, enough for conv2)
  for (int tt = wave; tt < 40; tt += 8) {
    int ct = tt & 7, mt = tt >> 3;
    v8f acc = (v8f){};
    for (int k = 0; k < nK; ++k) {
      const _Float16* wrow = wA + ((size_t)(ct * 16 + (lane & 15))) * (nK * nC) + k * nC;
      for (int cb = 0; cb < 4; ++cb) {
        v16h a;
        #pragma unroll
        for (int v = 0; v < 8; ++v) {
          int Ko = (v < 4) ? (h * 8 + 2 * v) : (16 + h * 8 + 2 * (v - 4));
          a[2 * v] = wrow[cb * 32 + Ko]; a[2 * v + 1] = wrow[cb * 32 + Ko + 1];
        }
        int trow = mt * 16 + n + k; if (trow > 71) trow = 71;
        const _Float16* xr = &Xs[trow * nC + cb * 32];
        v16h bf;
        #pragma unroll
        for (int v = 0; v < 8; ++v) {
          int Ko = (v < 4) ? (h * 8 + 2 * v) : (16 + h * 8 + 2 * (v - 4));
          bf[2 * v] = xr[Ko]; bf[2 * v + 1] = xr[Ko + 1];
        }
        acc = __builtin_amdgcn_wmma_f32_16x16x32_f16(false, a, false, bf,
                                                     (short)0, acc, false, false);
      }
    }
    #pragma unroll
    for (int i = 0; i < 8; i += 2) {
      float v0 = acc[i]     + b1[ct * 16 + h * 8 + i];
      float v1 = acc[i + 1] + b1[ct * 16 + h * 8 + i + 1];
      v0 = v0 > 0.f ? v0 : 0.01f * v0;
      v1 = v1 > 0.f ? v1 : 0.01f * v1;
      h2 p; p[0] = (_Float16)v0; p[1] = (_Float16)v1;
      *(h2*)&Ms[(mt * 16 + n) * nC + ct * 16 + h * 8 + i] = p;
    }
  }
  __syncthreads();

  // conv2 -> y (64 cols)
  for (int tt = wave; tt < 32; tt += 8) {
    int ct = tt & 7, mt = tt >> 3;
    v8f acc = (v8f){};
    for (int k = 0; k < nK; ++k) {
      const _Float16* wrow = wB + ((size_t)(ct * 16 + (lane & 15))) * (nK * nC) + k * nC;
      for (int cb = 0; cb < 4; ++cb) {
        v16h a;
        #pragma unroll
        for (int v = 0; v < 8; ++v) {
          int Ko = (v < 4) ? (h * 8 + 2 * v) : (16 + h * 8 + 2 * (v - 4));
          a[2 * v] = wrow[cb * 32 + Ko]; a[2 * v + 1] = wrow[cb * 32 + Ko + 1];
        }
        int mrow = mt * 16 + n + k;          // <= 67, in range
        const _Float16* xr = &Ms[mrow * nC + cb * 32];
        v16h bf;
        #pragma unroll
        for (int v = 0; v < 8; ++v) {
          int Ko = (v < 4) ? (h * 8 + 2 * v) : (16 + h * 8 + 2 * (v - 4));
          bf[2 * v] = xr[Ko]; bf[2 * v + 1] = xr[Ko + 1];
        }
        acc = __builtin_amdgcn_wmma_f32_16x16x32_f16(false, a, false, bf,
                                                     (short)0, acc, false, false);
      }
    }
    #pragma unroll
    for (int i = 0; i < 8; ++i) {
      float v = tanhf(acc[i] + b2[ct * 16 + h * 8 + i]);
      y[((size_t)b * nC + ct * 16 + h * 8 + i) * T + t0 + mt * 16 + n] = v;
    }
  }
}

// xt = xz * exp(y)   (+ edge-padded f16 copy)
__global__ void __launch_bounds__(256) k_mulexp(const float* __restrict__ xz,
                                                const float* __restrict__ y,
                                                float* __restrict__ xt,
                                                _Float16* __restrict__ xtp, int T) {
  int Tp = T + 8;
  size_t i = (size_t)blockIdx.x * 256 + threadIdx.x;
  if (i >= (size_t)nB * nC * Tp) return;
  int tp = (int)(i % Tp);
  size_t row = i / Tp;       // b*C + ci
  int t = tp - 4;
  t = t < 0 ? 0 : (t >= T ? T - 1 : t);
  size_t j = row * T + t;
  float v = xz[j] * __expf(y[j]);
  xtp[i] = (_Float16)v;
  if (tp >= 4 && tp < T + 4) xt[j] = v;
}

// xeu = xet + y2 ; xou = xot - y3.  Leaf: zip into strided dec view.
__global__ void __launch_bounds__(256) k_combine(const float* __restrict__ xet,
                                                 const float* __restrict__ xot,
                                                 const float* __restrict__ y2,
                                                 const float* __restrict__ y3,
                                                 float* __restrict__ dec, int off, int step,
                                                 float* __restrict__ ein,
                                                 float* __restrict__ oin, int T, int leaf) {
  size_t i = (size_t)blockIdx.x * 256 + threadIdx.x;
  if (i >= (size_t)nB * nC * T) return;
  int t = (int)(i % T);
  size_t t2 = i / T;
  int ci = (int)(t2 % nC);
  int b  = (int)(t2 / nC);
  float eu = xet[i] + y2[i];
  float ou = xot[i] - y3[i];
  if (leaf) {
    dec[((size_t)b * nL + off + (2 * t) * step)     * nC + ci] = eu;
    dec[((size_t)b * nL + off + (2 * t + 1) * step) * nC + ci] = ou;
  } else {
    ein[((size_t)b * T + t) * nC + ci] = eu;
    oin[((size_t)b * T + t) * nC + ci] = ou;
  }
}

// out[b][pi][c] = (sum_l projW[1024+pi][l] * (dec+x0)[b][l][c]) * std + mean, c<3
__global__ void __launch_bounds__(128) k_final(const float* __restrict__ dec,
                                               const float* __restrict__ x0,
                                               const float* __restrict__ projw,
                                               const float* __restrict__ means,
                                               const float* __restrict__ stdev,
                                               float* __restrict__ out) {
  __shared__ float red[3][4];
  int b  = blockIdx.x / nP;
  int pi = blockIdx.x % nP;
  int tid = threadIdx.x;
  const float* pw = projw + (size_t)(nL + pi) * nL;
  float s0 = 0.f, s1 = 0.f, s2 = 0.f;
  for (int l = tid; l < nL; l += 128) {
    float w = pw[l];
    const float* dp = dec + ((size_t)b * nL + l) * nC;
    const float* xp = x0  + ((size_t)b * nL + l) * nC;
    s0 += w * (dp[0] + xp[0]);
    s1 += w * (dp[1] + xp[1]);
    s2 += w * (dp[2] + xp[2]);
  }
  for (int off = 16; off; off >>= 1) {
    s0 += __shfl_xor(s0, off, 32);
    s1 += __shfl_xor(s1, off, 32);
    s2 += __shfl_xor(s2, off, 32);
  }
  if ((tid & 31) == 0) { red[0][tid >> 5] = s0; red[1][tid >> 5] = s1; red[2][tid >> 5] = s2; }
  __syncthreads();
  if (tid < 3) {
    float s = red[tid][0] + red[tid][1] + red[tid][2] + red[tid][3];
    out[((size_t)b * nP + pi) * 3 + tid] = s * stdev[b * nC + tid] + means[b * nC + tid];
  }
}

// ---------------------------------------------------------------------------
// Host-side tree recursion (pre-order node indices match reference counter)
// ---------------------------------------------------------------------------
namespace {

struct LevelBufs {
  float *xe, *xo, *xet, *xot, *y0, *y1, *ein, *oin;
  _Float16 *xep, *xop;
};

struct TreeCtx {
  LevelBufs LB[4];
  const _Float16 *cw1, *cw2;
  const float *cb1, *cb2;
  float* dec;
  hipStream_t stream;
  int counter;
};

void run_node(TreeCtx& cx, int level, const float* zin, int Lz, int off, int step) {
  const int node = cx.counter++;
  const int T = Lz / 2;
  LevelBufs& L = cx.LB[level];
  const size_t npad = (size_t)nB * nC * (T + 8);
  const size_t nbct = (size_t)nB * nC * T;
  const int gpad = (int)((npad + 255) / 256);
  const int gbct = (int)((nbct + 255) / 256);
  hipStream_t st = cx.stream;

  k_split<<<gpad, 256, 0, st>>>(zin, Lz, L.xe, L.xo, L.xep, L.xop, T);

  const size_t WF = (size_t)nC * nK * nC;
  const _Float16* w1n = cx.cw1 + (size_t)node * 4 * WF;
  const _Float16* w2n = cx.cw2 + (size_t)node * 4 * WF;
  const float* b1n = cx.cb1 + (size_t)node * 4 * nC;
  const float* b2n = cx.cb2 + (size_t)node * 4 * nC;
  dim3 gc(nB, T / 64);

  // y0 = ccb0(xo_pad);  y1 = ccb1(xe_pad)
  k_ccb<<<gc, 256, 0, st>>>(L.xop, w1n + 0 * WF, b1n + 0 * nC, w2n + 0 * WF, b2n + 0 * nC, L.y0, T, T + 8);
  k_ccb<<<gc, 256, 0, st>>>(L.xep, w1n + 1 * WF, b1n + 1 * nC, w2n + 1 * WF, b2n + 1 * nC, L.y1, T, T + 8);
  // xet = xe*exp(y0) (pads overwrite xep/xop);  xot = xo*exp(y1)
  k_mulexp<<<gpad, 256, 0, st>>>(L.xe, L.y0, L.xet, L.xep, T);
  k_mulexp<<<gpad, 256, 0, st>>>(L.xo, L.y1, L.xot, L.xop, T);
  // y2 = ccb2(xot_pad);  y3 = ccb3(xet_pad)
  k_ccb<<<gc, 256, 0, st>>>(L.xop, w1n + 2 * WF, b1n + 2 * nC, w2n + 2 * WF, b2n + 2 * nC, L.y0, T, T + 8);
  k_ccb<<<gc, 256, 0, st>>>(L.xep, w1n + 3 * WF, b1n + 3 * nC, w2n + 3 * WF, b2n + 3 * nC, L.y1, T, T + 8);

  int leaf = (level == 0) ? 1 : 0;
  k_combine<<<gbct, 256, 0, st>>>(L.xet, L.xot, L.y0, L.y1, cx.dec, off, step, L.ein, L.oin, T, leaf);

  if (!leaf) {
    run_node(cx, level - 1, L.ein, T, off, 2 * step);
    run_node(cx, level - 1, L.oin, T, off + step, 2 * step);
  }
}

} // namespace

// ---------------------------------------------------------------------------
// Orchestration
// ---------------------------------------------------------------------------
extern "C" void kernel_launch(void* const* d_in, const int* in_sizes, int n_in,
                              void* d_out, int out_size, void* d_ws, size_t ws_size,
                              hipStream_t stream) {
  (void)in_sizes; (void)n_in; (void)out_size; (void)ws_size;
  const float* x_enc = (const float*)d_in[0];
  const float* mask  = (const float*)d_in[1];
  const float* fc1w  = (const float*)d_in[2];   // (2048, 1024)
  const float* fc2w  = (const float*)d_in[3];   // (1024, 2048)
  const float* ln_g  = (const float*)d_in[4];
  const float* ln_b  = (const float*)d_in[5];
  const float* w1    = (const float*)d_in[6];   // (15,4,128,128,5)
  const float* b1    = (const float*)d_in[7];   // (15,4,128)
  const float* w2    = (const float*)d_in[8];
  const float* b2    = (const float*)d_in[9];
  const float* projw = (const float*)d_in[10];  // (1120, 1024)
  float* out = (float*)d_out;

  char* base = (char*)d_ws;
  size_t cur = 0;
  auto alloc = [&](size_t bytes) -> void* {
    void* p = base + cur;
    cur = (cur + bytes + 255) & ~(size_t)255;
    return p;
  };

  // --- persistent buffers ---
  float* means = (float*)alloc((size_t)nB * nC * 4);
  float* stdev = (float*)alloc((size_t)nB * nC * 4);
  float* x0    = (float*)alloc((size_t)nB * nL * nC * 4);
  float* dec   = (float*)alloc((size_t)nB * nL * nC * 4);
  const size_t WALL = (size_t)NNODES * 4 * nC * nK * nC;
  _Float16* cw1 = (_Float16*)alloc(WALL * 2);
  _Float16* cw2 = (_Float16*)alloc(WALL * 2);
  size_t persist_end = cur;

  // --- phase 1 scratch ---
  float*    xn    = (float*)alloc((size_t)nB * nL * nC * 4);
  _Float16* freqh = (_Float16*)alloc((size_t)nB * nC * nL * 2);
  _Float16* fc1h  = (_Float16*)alloc((size_t)2 * nL * nL * 2);
  _Float16* fc2h  = (_Float16*)alloc((size_t)2 * nL * nL * 2);
  _Float16* y1h   = (_Float16*)alloc((size_t)nB * nC * 2 * nL * 2);
  float*    lrr   = (float*)alloc((size_t)nB * nC * nL * 4);

  // --- phase 1 launches ---
  k_norm<<<(nB * nC) / 8, 256, 0, stream>>>(x_enc, mask, means, stdev);
  k_prep<<<(nB * nL * nC) / 256, 256, 0, stream>>>(x_enc, mask, means, stdev, xn);
  k_haar<<<nB * nC, 256, 0, stream>>>(xn, freqh);
  k_cvt<<<(2 * nL * nL + 255) / 256, 256, 0, stream>>>(fc1w, fc1h, 2 * nL * nL);
  k_cvt<<<(2 * nL * nL + 255) / 256, 256, 0, stream>>>(fc2w, fc2h, 2 * nL * nL);
  const int nw = NNODES * 4 * nC * nC * nK;
  k_cvtw<<<(nw + 255) / 256, 256, 0, stream>>>(w1, cw1, nw);
  k_cvtw<<<(nw + 255) / 256, 256, 0, stream>>>(w2, cw2, nw);
  k_gemm<0><<<dim3(32, 16), 256, 0, stream>>>(freqh, fc1h, (void*)y1h, nB * nC, 2 * nL, nL);
  k_gemm<1><<<dim3(32, 8),  256, 0, stream>>>(y1h, fc2h, (void*)lrr, nB * nC, nL, 2 * nL);
  k_ln<<<nB * nC, 256, 0, stream>>>(lrr, ln_g, ln_b);
  k_apply<<<(nB * nL * nC) / 256, 256, 0, stream>>>(xn, lrr, x0);

  // --- phase 2: reuse phase-1 scratch for tree buffers (stream-ordered) ---
  cur = persist_end;
  TreeCtx cx;
  cx.cw1 = cw1; cx.cw2 = cw2; cx.cb1 = b1; cx.cb2 = b2;
  cx.dec = dec; cx.stream = stream; cx.counter = 0;
  for (int lev = 3; lev >= 0; --lev) {
    int T = 64 << lev;
    LevelBufs& L = cx.LB[lev];
    L.xe  = (float*)alloc((size_t)nB * nC * T * 4);
    L.xo  = (float*)alloc((size_t)nB * nC * T * 4);
    L.xet = (float*)alloc((size_t)nB * nC * T * 4);
    L.xot = (float*)alloc((size_t)nB * nC * T * 4);
    L.y0  = (float*)alloc((size_t)nB * nC * T * 4);
    L.y1  = (float*)alloc((size_t)nB * nC * T * 4);
    L.xep = (_Float16*)alloc((size_t)nB * nC * (T + 8) * 2);
    L.xop = (_Float16*)alloc((size_t)nB * nC * (T + 8) * 2);
    if (lev > 0) {
      L.ein = (float*)alloc((size_t)nB * T * nC * 4);
      L.oin = (float*)alloc((size_t)nB * T * nC * 4);
    } else {
      L.ein = nullptr; L.oin = nullptr;
    }
  }
  run_node(cx, 3, x0, nL, 0, 1);

  // --- final: dec+x, 96x3 projection slice, de-normalize ---
  k_final<<<nB * nP, 128, 0, stream>>>(dec, x0, projw, means, stdev, out);
}
